// EGT_GRC_41394894799301
// MI455X (gfx1250) — compile-verified, matrix operand order checked by reading, not run on hardware
//
#include <hip/hip_runtime.h>
#include <hip/hip_bf16.h>

// ---------------------------------------------------------------------------
// Types for CDNA5 WMMA
// ---------------------------------------------------------------------------
typedef __attribute__((ext_vector_type(16))) __bf16   v16bf;
typedef __attribute__((ext_vector_type(8)))  float    v8f;
typedef __attribute__((ext_vector_type(4)))  unsigned v4u;

// ---------------------------------------------------------------------------
// Small helpers
// ---------------------------------------------------------------------------
__device__ inline float gelu_f(float x) {
    // tanh approximation (jax.nn.gelu default, approximate=True)
    const float kc = 0.7978845608028654f; // sqrt(2/pi)
    float x3 = x * x * x;
    return 0.5f * x * (1.0f + tanhf(kc * (x + 0.044715f * x3)));
}
__device__ inline float sigm_f(float x) { return 1.0f / (1.0f + expf(-x)); }

__device__ inline __bf16 f2bf(float f) {
    union { float f; unsigned u; } in; in.f = f;
    unsigned u = in.u;
    unsigned r = u + 0x7FFFu + ((u >> 16) & 1u);   // round-to-nearest-even
    union { unsigned short s; __bf16 b; } out;
    out.s = (unsigned short)(r >> 16);
    return out.b;
}

// block reduction, blockDim.x == 128
__device__ inline float block_reduce_sum128(float v, float* sbuf) {
    int tid = threadIdx.x;
    sbuf[tid] = v;
    __syncthreads();
    for (int off = 64; off > 0; off >>= 1) {
        if (tid < off) sbuf[tid] += sbuf[tid + off];
        __syncthreads();
    }
    float r = sbuf[0];
    __syncthreads();
    return r;
}

// ---------------------------------------------------------------------------
// fp32 -> bf16 elementwise
// ---------------------------------------------------------------------------
__global__ void f32_to_bf16_kernel(const float* __restrict__ x,
                                   __bf16* __restrict__ y, int n) {
    int i = blockIdx.x * blockDim.x + threadIdx.x;
    if (i < n) y[i] = f2bf(x[i]);
}

// ---------------------------------------------------------------------------
// Pack row-major fp32 B(K x N) into fragment-major bf16 layout for
// V_WMMA_F32_16X16X32_BF16.  Fragment = 32x16 (KxN) tile = 512 elems.
// Fragment index = no * (K/32) + ko.  Within fragment: lane-major,
// lane L owns 16 contiguous bf16.  Per the ISA 16-bit B layout:
//   lane L: n = n0 + (L&15), half = L>>4
//   element e (0..7)  -> k = k0 + 8*half + e
//   element e (8..15) -> k = k0 + 16 + 8*half + (e-8)
// so a lane's fragment loads as two contiguous b128s.
// ---------------------------------------------------------------------------
__global__ void pack_B_kernel(const float* __restrict__ B,
                              __bf16* __restrict__ out, int K, int N) {
    size_t idx = (size_t)blockIdx.x * blockDim.x + threadIdx.x;
    size_t tot = (size_t)K * N;
    if (idx >= tot) return;
    size_t frag = idx >> 9;          // /512
    int within = (int)(idx & 511);
    int lane = within >> 4;
    int e    = within & 15;
    int nK32 = K >> 5;
    int no = (int)(frag / nK32);
    int ko = (int)(frag % nK32);
    int half = lane >> 4;
    int n = (no << 4) + (lane & 15);
    int k = (ko << 5) + ((e < 8) ? (8 * half + e) : (16 + 8 * half + (e - 8)));
    out[idx] = f2bf(B[(size_t)k * N + n]);
}

// ---------------------------------------------------------------------------
// Load a 16x32 bf16 A fragment from row-major A (lda = K), per ISA layout.
// lane L: m = row0 + (L&15), half = L>>4
//   elements 0..7  <- A[m][k0 + 8*half .. +7]          (one b128)
//   elements 8..15 <- A[m][k0 + 16 + 8*half .. +7]     (one b128)
// ---------------------------------------------------------------------------
__device__ inline v16bf load_frag_A(const __bf16* __restrict__ A, int lda,
                                    int row0, int k0) {
    int lane = threadIdx.x & 31;
    int m = lane & 15, half = lane >> 4;
    const __bf16* p = A + (size_t)(row0 + m) * lda + k0 + 8 * half;
    union { v4u u[2]; v16bf b; } u;
    u.u[0] = *(const v4u*)(p);
    u.u[1] = *(const v4u*)(p + 16);
    return u.b;
}

// ---------------------------------------------------------------------------
// bf16 WMMA GEMM: C(M x N) = A(M x K) @ B(K x N) + bias[N]
//   A: row-major bf16.  Bp: fragment-packed bf16 (pack_B_kernel).
//   ACT==0: C fp32 out.   ACT==1: gelu, bf16 out (for the H intermediate).
// One wave computes a 16 x 64 tile (4 accumulators, A-frag reuse x4).
// blockDim.x = 128 (4 waves).  M%16==0, N%64==0, K%32==0.
// ---------------------------------------------------------------------------
template <int ACT>
__global__ void gemm_wmma_kernel(const __bf16* __restrict__ A,
                                 const __bf16* __restrict__ Bp,
                                 const float* __restrict__ bias,
                                 float* __restrict__ C,
                                 __bf16* __restrict__ Cbf,
                                 int M, int N, int K) {
    int wave = blockIdx.x * 4 + (threadIdx.x >> 5);
    int nT64 = N >> 6;
    int mt = wave / nT64;
    int nt = wave % nT64;
    if (mt * 16 >= M) return;              // wave-uniform: EXEC stays all-ones

    int lane = threadIdx.x & 31;
    int nK32 = K >> 5;

    v8f acc0 = {0.f,0.f,0.f,0.f,0.f,0.f,0.f,0.f};
    v8f acc1 = acc0, acc2 = acc0, acc3 = acc0;

    const __bf16* bbase = Bp + (((size_t)nt * 4) * nK32 << 9) + (size_t)lane * 16;
    for (int ko = 0; ko < nK32; ++ko) {
        v16bf a = load_frag_A(A, K, mt * 16, ko * 32);
#pragma unroll
        for (int t = 0; t < 4; ++t) {
            const __bf16* bp = bbase + (((size_t)t * nK32 + ko) << 9);
            union { v4u u[2]; v16bf b; } ub;
            ub.u[0] = *(const v4u*)(bp);
            ub.u[1] = *(const v4u*)(bp + 8);
            v8f cin = (t == 0) ? acc0 : (t == 1) ? acc1 : (t == 2) ? acc2 : acc3;
            v8f cout = __builtin_amdgcn_wmma_f32_16x16x32_bf16(
                false, a, false, ub.b, (short)0, cin, false, false);
            if      (t == 0) acc0 = cout;
            else if (t == 1) acc1 = cout;
            else if (t == 2) acc2 = cout;
            else             acc3 = cout;
        }
    }

    // C/D layout: vgpr r, lane L -> m = r + 8*(L>>4), n = (L&15)
    int m_l = (lane >> 4) * 8;
    int n_l = lane & 15;
#pragma unroll
    for (int t = 0; t < 4; ++t) {
        v8f acc = (t == 0) ? acc0 : (t == 1) ? acc1 : (t == 2) ? acc2 : acc3;
        int nb = nt * 64 + t * 16 + n_l;
        float bsv = bias[nb];
#pragma unroll
        for (int r = 0; r < 8; ++r) {
            int m = mt * 16 + m_l + r;
            float v = acc[r] + bsv;
            if (ACT == 1) Cbf[(size_t)m * N + nb] = f2bf(gelu_f(v));
            else          C[(size_t)m * N + nb] = v;
        }
    }
}

// ---------------------------------------------------------------------------
// Row LayerNorm: y[row,:] = LN(x[row,:]) * g + b ; D == 512, blockDim = 128
// ---------------------------------------------------------------------------
__global__ void ln_rows_kernel(const float* __restrict__ x,
                               const float* __restrict__ g,
                               const float* __restrict__ b,
                               float* __restrict__ y, int D) {
    __shared__ float sred[128];
    int row = blockIdx.x, tid = threadIdx.x;
    const float* xr = x + (size_t)row * D;
    float v[4];
#pragma unroll
    for (int j = 0; j < 4; ++j) v[j] = xr[tid + j * 128];
    float mean = block_reduce_sum128(v[0] + v[1] + v[2] + v[3], sred) / (float)D;
    float vs = 0.f;
#pragma unroll
    for (int j = 0; j < 4; ++j) { float d = v[j] - mean; vs += d * d; }
    float var = block_reduce_sum128(vs, sred) / (float)D;
    float inv = rsqrtf(var + 1e-5f);
    float* yr = y + (size_t)row * D;
#pragma unroll
    for (int j = 0; j < 4; ++j) {
        int d = tid + j * 128;
        yr[d] = (v[j] - mean) * inv * g[d] + b[d];
    }
}

// ---------------------------------------------------------------------------
// Scorer: logits[n,s] = W_d2 . gelu(W_d1^T feats + b_d1) + b_d2  (masked)
// feats = [cs[n,s,0:64], cs[n,s+1,0:64]],  cs shape (N, Scur, D) compact.
// One thread per (n,s); W_d1 (128x64) staged in LDS.
// ---------------------------------------------------------------------------
__global__ void scorer_kernel(const float* __restrict__ cs,
                              const float* __restrict__ mask,
                              const float* __restrict__ Wd1,
                              const float* __restrict__ bd1,
                              const float* __restrict__ Wd2,
                              const float* __restrict__ bd2,
                              float* __restrict__ logits,
                              int Nb, int Scur, int D, int i_off, int S0) {
    __shared__ float sW[128 * 64];
    int tid = threadIdx.x;
    for (int j = tid; j < 128 * 64; j += blockDim.x) sW[j] = Wd1[j];
    __syncthreads();

    int L = Scur - 1;
    int idx = blockIdx.x * blockDim.x + tid;
    if (idx >= Nb * L) return;
    int n = idx / L, s = idx % L;
    const float* lrow = cs + ((size_t)n * Scur + s) * D;

    float h[64];
#pragma unroll
    for (int j = 0; j < 64; ++j) h[j] = bd1[j];
    for (int t = 0; t < 128; ++t) {
        float f = (t < 64) ? lrow[t] : lrow[D + (t - 64)];
#pragma unroll
        for (int j = 0; j < 64; ++j) h[j] = fmaf(f, sW[t * 64 + j], h[j]);
    }
    float acc = bd2[0];
#pragma unroll
    for (int j = 0; j < 64; ++j) acc += Wd2[j] * gelu_f(h[j]);

    float m = mask[(size_t)n * S0 + i_off + s];
    logits[idx] = (m > 0.f) ? acc : -1e9f;
}

// argmax over s (first occurrence, matching jnp.argmax)
__global__ void argmax_kernel(const float* __restrict__ logits,
                              int* __restrict__ kidx, int Nb, int L) {
    int n = blockIdx.x * blockDim.x + threadIdx.x;
    if (n >= Nb) return;
    const float* row = logits + (size_t)n * L;
    float best = row[0]; int bi = 0;
    for (int s = 1; s < L; ++s) {
        float v = row[s];
        if (v > best) { best = v; bi = s; }
    }
    kidx[n] = bi;
}

__global__ void setk0_kernel(int* __restrict__ kidx, int Nb) {
    int n = blockIdx.x * blockDim.x + threadIdx.x;
    if (n < Nb) kidx[n] = 0;
}

// gather l_sel/r_sel (fp32) and their bf16 concat (GEMM-A input)
__global__ void gather_kernel(const float* __restrict__ cs,
                              const int* __restrict__ kidx,
                              float* __restrict__ lsel,
                              float* __restrict__ rsel,
                              __bf16* __restrict__ lr,
                              int Nb, int Scur, int D) {
    int idx = blockIdx.x * blockDim.x + threadIdx.x;
    if (idx >= Nb * D) return;
    int n = idx / D, d = idx % D;
    int k = kidx[n];
    float lv = cs[((size_t)n * Scur + k) * D + d];
    float rv = cs[((size_t)n * Scur + k + 1) * D + d];
    lsel[idx] = lv;
    rsel[idx] = rv;
    lr[(size_t)n * 2 * D + d]     = f2bf(lv);
    lr[(size_t)n * 2 * D + D + d] = f2bf(rv);
}

// gated combine + LayerNorm; one block (128 thr) per n; D == 512
__global__ void grc_combine_kernel(const float* __restrict__ contents,
                                   const float* __restrict__ lsel,
                                   const float* __restrict__ rsel,
                                   const float* __restrict__ g,
                                   const float* __restrict__ beta,
                                   float* __restrict__ outv, int D) {
    __shared__ float sred[128];
    int n = blockIdx.x, tid = threadIdx.x;
    const float* crow = contents + (size_t)n * 4 * D;
    float sv[4];
#pragma unroll
    for (int j = 0; j < 4; ++j) {
        int d = tid + j * 128;
        float c0 = crow[0 * D + d];
        float c1 = crow[1 * D + d];
        float c2 = crow[2 * D + d];
        float c3 = crow[3 * D + d];
        float li = lsel[(size_t)n * D + d];
        float ri = rsel[(size_t)n * D + d];
        sv[j] = sigm_f(c0) * li + sigm_f(c1) * ri + sigm_f(c2) * c3;
    }
    float mean = block_reduce_sum128(sv[0] + sv[1] + sv[2] + sv[3], sred) / (float)D;
    float vs = 0.f;
#pragma unroll
    for (int j = 0; j < 4; ++j) { float d = sv[j] - mean; vs += d * d; }
    float var = block_reduce_sum128(vs, sred) / (float)D;
    float inv = rsqrtf(var + 1e-5f);
#pragma unroll
    for (int j = 0; j < 4; ++j) {
        int d = tid + j * 128;
        outv[(size_t)n * D + d] = (sv[j] - mean) * inv * g[d] + beta[d];
    }
}

// scatter/shrink: nxt(N, L, D) from cur(N, Scur, D), winner row replaced
__global__ void update_kernel(const float* __restrict__ cur,
                              const float* __restrict__ newv,
                              const int* __restrict__ kidx,
                              const float* __restrict__ mask,
                              float* __restrict__ nxt,
                              int Nb, int Scur, int D, int ip1, int S0) {
    int L = Scur - 1;
    size_t idx = (size_t)blockIdx.x * blockDim.x + threadIdx.x;
    size_t tot = (size_t)Nb * L * D;
    if (idx >= tot) return;
    int d = (int)(idx % D);
    size_t t = idx / D;
    int s = (int)(t % L);
    int n = (int)(t / L);
    float done = mask[(size_t)n * S0 + ip1];
    int k = kidx[n];
    float v;
    if (s < k)       v = cur[((size_t)n * Scur + s) * D + d];
    else if (s == k) v = newv[(size_t)n * D + d];
    else             v = cur[((size_t)n * Scur + s + 1) * D + d];
    float keep = cur[((size_t)n * Scur + s) * D + d];
    nxt[idx] = done * v + (1.0f - done) * keep;
}

// ---------------------------------------------------------------------------
// Host side
// ---------------------------------------------------------------------------
template <int ACT>
static inline void launch_gemm(const __bf16* A, const __bf16* Bp,
                               const float* bias, float* C, __bf16* Cbf,
                               int M, int K, int N, hipStream_t stream) {
    int waves = (M / 16) * (N / 64);
    int blocks = (waves + 3) / 4;
    gemm_wmma_kernel<ACT><<<blocks, 128, 0, stream>>>(A, Bp, bias, C, Cbf, M, N, K);
}

extern "C" void kernel_launch(void* const* d_in, const int* in_sizes, int n_in,
                              void* d_out, int out_size, void* d_ws, size_t ws_size,
                              hipStream_t stream) {
    const int N = 128, S0 = 64, D = 512, CH = 2048;

    const float* input      = (const float*)d_in[0];
    const float* input_mask = (const float*)d_in[1];
    const float* W_init     = (const float*)d_in[2];
    const float* b_init     = (const float*)d_in[3];
    const float* g_init     = (const float*)d_in[4];
    const float* beta_init  = (const float*)d_in[5];
    const float* W_d1       = (const float*)d_in[6];
    const float* b_d1       = (const float*)d_in[7];
    const float* W_d2       = (const float*)d_in[8];
    const float* b_d2       = (const float*)d_in[9];
    const float* W_c1       = (const float*)d_in[10];
    const float* b_c1       = (const float*)d_in[11];
    const float* W_c2       = (const float*)d_in[12];
    const float* b_c2       = (const float*)d_in[13];
    const float* g_c        = (const float*)d_in[14];
    const float* beta_c     = (const float*)d_in[15];

    char* cur_ws = (char*)d_ws;
    auto walloc = [&](size_t bytes) -> char* {
        char* p = cur_ws;
        cur_ws += (bytes + 255) & ~(size_t)255;
        return p;
    };

    __bf16* inp_bf   = (__bf16*)walloc((size_t)N * S0 * D * 2);       // 8 MB
    __bf16* WiP      = (__bf16*)walloc((size_t)D * D * 2);            // 0.5 MB
    __bf16* Wc1P     = (__bf16*)walloc((size_t)2 * D * CH * 2);       // 4 MB
    __bf16* Wc2P     = (__bf16*)walloc((size_t)CH * 4 * D * 2);       // 8 MB
    float*  csA      = (float*)walloc((size_t)N * S0 * D * 4);        // 16 MB
    float*  csB      = (float*)walloc((size_t)N * S0 * D * 4);        // 16 MB
    float*  logits   = (float*)walloc((size_t)N * (S0 - 1) * 4);
    int*    kidx     = (int*)walloc((size_t)N * 4);
    float*  lsel     = (float*)walloc((size_t)N * D * 4);
    float*  rsel     = (float*)walloc((size_t)N * D * 4);
    __bf16* lr_bf    = (__bf16*)walloc((size_t)N * 2 * D * 2);
    __bf16* H_bf     = (__bf16*)walloc((size_t)N * CH * 2);
    float*  contents = (float*)walloc((size_t)N * 4 * D * 4);
    float*  newv     = (float*)walloc((size_t)N * D * 4);

    // --- one-time conversions / weight packing ---
    {
        int n = N * S0 * D;
        f32_to_bf16_kernel<<<(n + 255) / 256, 256, 0, stream>>>(input, inp_bf, n);
    }
    {
        size_t n = (size_t)D * D;
        pack_B_kernel<<<(int)((n + 255) / 256), 256, 0, stream>>>(W_init, WiP, D, D);
    }
    {
        size_t n = (size_t)2 * D * CH;
        pack_B_kernel<<<(int)((n + 255) / 256), 256, 0, stream>>>(W_c1, Wc1P, 2 * D, CH);
    }
    {
        size_t n = (size_t)CH * 4 * D;
        pack_B_kernel<<<(int)((n + 255) / 256), 256, 0, stream>>>(W_c2, Wc2P, CH, 4 * D);
    }

    // --- cs = LN(input @ W_init + b_init) ---
    launch_gemm<0>(inp_bf, WiP, b_init, csB /*temp*/, nullptr,
                   N * S0, D, D, stream);
    ln_rows_kernel<<<N * S0, 128, 0, stream>>>(csB, g_init, beta_init, csA, D);

    float* curp = csA;
    float* nxtp = csB;

    for (int i = 0; i < S0 - 1; ++i) {
        int Scur = S0 - i;
        int L = Scur - 1;

        if (i < S0 - 2) {
            int tot = N * L;
            scorer_kernel<<<(tot + 127) / 128, 128, 0, stream>>>(
                curp, input_mask, W_d1, b_d1, W_d2, b_d2, logits,
                N, Scur, D, i + 1, S0);
            argmax_kernel<<<1, 128, 0, stream>>>(logits, kidx, N, L);
        } else {
            setk0_kernel<<<1, 128, 0, stream>>>(kidx, N);
        }

        gather_kernel<<<(N * D + 255) / 256, 256, 0, stream>>>(
            curp, kidx, lsel, rsel, lr_bf, N, Scur, D);

        // H = gelu(lr @ W_c1 + b_c1)   (bf16 out)
        launch_gemm<1>(lr_bf, Wc1P, b_c1, nullptr, H_bf, N, 2 * D, CH, stream);
        // contents = H @ W_c2 + b_c2   (fp32 out)
        launch_gemm<0>(H_bf, Wc2P, b_c2, contents, nullptr, N, CH, 4 * D, stream);

        grc_combine_kernel<<<N, 128, 0, stream>>>(contents, lsel, rsel,
                                                  g_c, beta_c, newv, D);

        size_t tot = (size_t)N * L * D;
        update_kernel<<<(int)((tot + 255) / 256), 256, 0, stream>>>(
            curp, newv, kidx, input_mask, nxtp, N, Scur, D, i + 1, S0);

        float* tmp = curp; curp = nxtp; nxtp = tmp;
    }

    // After 63 reductions cs is (N, 1, D) compact == N*D contiguous floats.
    hipMemcpyAsync(d_out, curp, (size_t)N * D * sizeof(float),
                   hipMemcpyDeviceToDevice, stream);
}
